// SpectralGroupedQueryAttention_66468913873566
// MI455X (gfx1250) — matrix operand order repaired and checked by reading, not compiled, not run on hardware
//
#include <hip/hip_runtime.h>

typedef __bf16 bf16;
typedef __attribute__((ext_vector_type(16))) __bf16 v16bf;
typedef __attribute__((ext_vector_type(8)))  __bf16 v8bf;
typedef __attribute__((ext_vector_type(8)))  float  v8f;

#define HW      65536
#define IMGW    256
#define CDIM    192
#define CKV     96
#define CK      48
#define NB      4
#define KSPLIT  32

// ---------- WMMA fragment helpers (ISA 16-bit A 16x32 / B 32x16 layouts) ----------
__device__ __forceinline__ v16bf load_a_frag(const bf16* rowK, int hi) {
    // element j<8 -> K = hi+j ; j>=8 -> K = 16+hi+(j-8)   (hi = lane>=16 ? 8 : 0)
    union { v16bf v; v8bf h[2]; } u;
    u.h[0] = *(const v8bf*)(rowK + hi);
    u.h[1] = *(const v8bf*)(rowK + 16 + hi);
    return u.v;
}
__device__ __forceinline__ v16bf load_b_frag(const bf16* rowK, int k0b) {
    // element j<8 -> K = k0b/2+j ; j>=8 -> K = 16+k0b/2+(j-8)   (k0b = lane>=16 ? 16 : 0)
    union { v16bf v; v8bf h[2]; } u;
    u.h[0] = *(const v8bf*)(rowK + (k0b >> 1));
    u.h[1] = *(const v8bf*)(rowK + 16 + (k0b >> 1));
    return u.v;
}
__device__ __forceinline__ v16bf zero_frag() {
    v16bf z;
#pragma unroll
    for (int i = 0; i < 16; ++i) z[i] = (bf16)0.0f;
    return z;
}
__device__ __forceinline__ v8f zero_acc() {
    v8f z;
#pragma unroll
    for (int i = 0; i < 8; ++i) z[i] = 0.0f;
    return z;
}

// ---------- small elementwise kernels ----------
__global__ void cvt_bf_kernel(const float* __restrict__ in, bf16* __restrict__ out, int n) {
    int i = blockIdx.x * 256 + threadIdx.x;
    if (i < n) out[i] = (bf16)in[i];
}

__global__ void zerof_kernel(float* __restrict__ p, int n) {
    int i = blockIdx.x * 256 + threadIdx.x;
    if (i < n) p[i] = 0.0f;
}

// LayerNorm over channels (channel-first f32 in) -> channel-last bf16 out
__global__ void ln_kernel(const float* __restrict__ x, const float* __restrict__ lw,
                          const float* __restrict__ lb, bf16* __restrict__ xt) {
    int idx = blockIdx.x * 256 + threadIdx.x;         // NB*HW pixels
    int b = idx / HW, p = idx % HW;
    const float* xp = x + (size_t)b * CDIM * HW + p;
    float s = 0.f, s2 = 0.f;
    for (int c = 0; c < CDIM; ++c) { float v = xp[(size_t)c * HW]; s += v; s2 += v * v; }
    float mu  = s * (1.0f / CDIM);
    float var = s2 * (1.0f / CDIM) - mu * mu;
    float rs  = rsqrtf(var + 1e-5f);
    bf16* o = xt + (size_t)idx * CDIM;
    for (int c = 0; c < CDIM; ++c) {
        float v = xp[(size_t)c * HW];
        o[c] = (bf16)(((v - mu) * rs) * lw[c] + lb[c]);
    }
}

// depthwise 3x3, SAME; channel-last bf16 in/out
__global__ void dwconv_kernel(const bf16* __restrict__ in, const float* __restrict__ w,
                              bf16* __restrict__ out, int Cc) {
    long idx = (long)blockIdx.x * 256 + threadIdx.x;  // NB*HW*Cc
    int  c  = (int)(idx % Cc);
    long bp = idx / Cc;
    int  p  = (int)(bp % HW);
    int  b  = (int)(bp / HW);
    int  py = p >> 8, px = p & 255;
    float acc = 0.f;
#pragma unroll
    for (int ky = 0; ky < 3; ++ky) {
        int yy = py + ky - 1;
        if ((unsigned)yy >= (unsigned)IMGW) continue;
#pragma unroll
        for (int kx = 0; kx < 3; ++kx) {
            int xx = px + kx - 1;
            if ((unsigned)xx >= (unsigned)IMGW) continue;
            acc += (float)in[((size_t)b * HW + yy * IMGW + xx) * Cc + c] * w[c * 9 + ky * 3 + kx];
        }
    }
    out[idx] = (bf16)acc;
}

// channel-last [b,HW,Cc] -> channel-first [b,Cc,HW], 32x32 LDS tiles
__global__ void transpose_kernel(const bf16* __restrict__ in, bf16* __restrict__ out, int Cc) {
    __shared__ unsigned short sm[32][33];
    int tiles_c     = Cc >> 5;
    int tiles_per_b = (HW >> 5) * tiles_c;
    int t = blockIdx.x;
    int b = t / tiles_per_b, r = t % tiles_per_b;
    int pt = r / tiles_c, ct = r % tiles_c;
    const unsigned short* src = (const unsigned short*)(in + ((size_t)b * HW + pt * 32) * Cc + ct * 32);
#pragma unroll
    for (int i = 0; i < 4; ++i) {
        int q = threadIdx.x + i * 256;
        sm[q >> 5][q & 31] = src[(size_t)(q >> 5) * Cc + (q & 31)];
    }
    __syncthreads();
    unsigned short* dst = (unsigned short*)(out + ((size_t)b * Cc + ct * 32) * HW + pt * 32);
#pragma unroll
    for (int i = 0; i < 4; ++i) {
        int q = threadIdx.x + i * 256;
        dst[(size_t)(q >> 5) * HW + (q & 31)] = sm[q & 31][q >> 5];
    }
}

// 1/||row||_2 for selected channels; channel-first bf16 [b, chan_tot, HW]
__global__ void rownorm_kernel(const bf16* __restrict__ in, float* __restrict__ rn,
                               int chan_sel, int chan_tot) {
    int rid = blockIdx.x;
    int b = rid / chan_sel, cs = rid % chan_sel;
    const bf16* row = in + ((size_t)b * chan_tot + cs) * HW;
    float s = 0.f;
    for (int i = threadIdx.x; i < HW; i += 256) { float v = (float)row[i]; s += v * v; }
    __shared__ float sm[256];
    sm[threadIdx.x] = s; __syncthreads();
    for (int o = 128; o > 0; o >>= 1) {
        if (threadIdx.x < o) sm[threadIdx.x] += sm[threadIdx.x + o];
        __syncthreads();
    }
    if (threadIdx.x == 0) rn[rid] = 1.0f / fmaxf(sqrtf(sm[0]), 1e-12f);
}

__global__ void scale_rows_kernel(bf16* __restrict__ buf, const float* __restrict__ rn,
                                  int chan_sel, int chan_tot) {
    long idx = (long)blockIdx.x * 256 + threadIdx.x;  // NB*chan_sel*HW
    int rid = (int)(idx / HW);
    int b = rid / chan_sel, cs = rid % chan_sel;
    size_t a = ((size_t)b * chan_tot + cs) * HW + (size_t)(idx % HW);
    buf[a] = (bf16)((float)buf[a] * rn[rid]);
}

// ---------- WMMA GEMM (2x2 register-blocked): forward conv1x1 ----------
// D[p,o] = sum_c X[p,c] * W[o,c] ; X channel-last bf16 [NB*HW, 192], W [Nout, 192]
__global__ void gemm_fwd_kernel(const bf16* __restrict__ X, const bf16* __restrict__ W,
                                bf16* __restrict__ O, int Nout, int nblks) {
    int w   = blockIdx.x * 8 + (threadIdx.x >> 5);   // wave over (mblk, nblk)
    int l   = threadIdx.x & 31;
    int mb  = w / nblks, nb = w % nblks;
    int lm  = l & 15;
    int hi  = (l >= 16) ? 8 : 0;
    int k0b = (l >= 16) ? 16 : 0;
    const bf16* Arow[2];
    const bf16* Brow[2];
#pragma unroll
    for (int t = 0; t < 2; ++t) {
        Arow[t] = X + (size_t)(mb * 32 + t * 16 + lm) * CDIM;
        Brow[t] = W + (size_t)(nb * 32 + t * 16 + lm) * CDIM;
    }
    v8f acc[2][2];
#pragma unroll
    for (int i = 0; i < 2; ++i)
#pragma unroll
        for (int j = 0; j < 2; ++j) acc[i][j] = zero_acc();
#pragma unroll
    for (int k = 0; k < CDIM; k += 32) {
        v16bf a[2], b[2];
#pragma unroll
        for (int t = 0; t < 2; ++t) {
            a[t] = load_a_frag(Arow[t] + k, hi);
            b[t] = load_b_frag(Brow[t] + k, k0b);
        }
#pragma unroll
        for (int i = 0; i < 2; ++i)
#pragma unroll
            for (int j = 0; j < 2; ++j)
                acc[i][j] = __builtin_amdgcn_wmma_f32_16x16x32_bf16(
                    false, a[i], false, b[j], (short)0, acc[i][j], false, false);
    }
#pragma unroll
    for (int i = 0; i < 2; ++i)
#pragma unroll
        for (int j = 0; j < 2; ++j) {
            bf16* dst = O + (size_t)(mb * 32 + i * 16 + hi) * Nout + nb * 32 + j * 16 + lm;
#pragma unroll
            for (int r = 0; r < 8; ++r) dst[(size_t)r * Nout] = (bf16)acc[i][j][r];
        }
}

// ---------- WMMA GEMM (2x2 blocked): proj -> f32 channel-first d_out ----------
// d_out[b,o,p] = sum_c W[o,c] * X[b*HW+p, c]
__global__ void gemm_proj_kernel(const bf16* __restrict__ Wp, const bf16* __restrict__ X,
                                 float* __restrict__ O) {
    int w   = blockIdx.x * 8 + (threadIdx.x >> 5);
    int l   = threadIdx.x & 31;
    int mb  = w % (CDIM / 32);      // over out-channel blocks (6)
    int nb  = w / (CDIM / 32);      // over global pixel blocks (NB*HW/32)
    int lm  = l & 15;
    int hi  = (l >= 16) ? 8 : 0;
    int k0b = (l >= 16) ? 16 : 0;
    const bf16* Arow[2];
    const bf16* Brow[2];
#pragma unroll
    for (int t = 0; t < 2; ++t) {
        Arow[t] = Wp + (size_t)(mb * 32 + t * 16 + lm) * CDIM;
        Brow[t] = X + (size_t)(nb * 32 + t * 16 + lm) * CDIM;
    }
    v8f acc[2][2];
#pragma unroll
    for (int i = 0; i < 2; ++i)
#pragma unroll
        for (int j = 0; j < 2; ++j) acc[i][j] = zero_acc();
#pragma unroll
    for (int k = 0; k < CDIM; k += 32) {
        v16bf a[2], b[2];
#pragma unroll
        for (int t = 0; t < 2; ++t) {
            a[t] = load_a_frag(Arow[t] + k, hi);
            b[t] = load_b_frag(Brow[t] + k, k0b);
        }
#pragma unroll
        for (int i = 0; i < 2; ++i)
#pragma unroll
            for (int j = 0; j < 2; ++j)
                acc[i][j] = __builtin_amdgcn_wmma_f32_16x16x32_bf16(
                    false, a[i], false, b[j], (short)0, acc[i][j], false, false);
    }
    int bb = (nb * 32) / HW;
    int p0 = (nb * 32) % HW;
#pragma unroll
    for (int i = 0; i < 2; ++i)
#pragma unroll
        for (int j = 0; j < 2; ++j) {
            float* dst = O + (size_t)bb * CDIM * HW + (size_t)(mb * 32 + i * 16 + hi) * HW
                           + p0 + j * 16 + lm;
#pragma unroll
            for (int r = 0; r < 8; ++r) dst[(size_t)r * HW] = acc[i][j][r];
        }
}

// ---------- WMMA q.k^T, split-K over pixels, f32 atomics into 32x32-padded attn ----------
__global__ void qk_kernel(const bf16* __restrict__ Q, const bf16* __restrict__ Kc,
                          float* __restrict__ attn) {
    int bh  = blockIdx.x / KSPLIT;
    int ks  = blockIdx.x % KSPLIT;
    int wid = threadIdx.x >> 5;
    int l   = threadIdx.x & 31;
    int b = bh >> 3, h = bh & 7, g = h >> 2;
    int lm  = l & 15;
    int hi  = (l >= 16) ? 8 : 0;
    int k0b = (l >= 16) ? 16 : 0;

    const bf16* Arow[2]; bool av[2];
    const bf16* Brow[2]; bool bv[2];
#pragma unroll
    for (int t = 0; t < 2; ++t) {
        int m = t * 16 + lm;
        av[t] = m < 24;
        Arow[t] = Q + ((size_t)b * CDIM + h * 24 + (av[t] ? m : 0)) * HW;
        bv[t] = m < 24;
        Brow[t] = Kc + ((size_t)b * CKV + g * 24 + (bv[t] ? m : 0)) * HW;
    }
    v8f acc[2][2];
#pragma unroll
    for (int i = 0; i < 2; ++i)
#pragma unroll
        for (int j = 0; j < 2; ++j) acc[i][j] = zero_acc();

    const int steps = (HW / 32) / KSPLIT;   // 64 k-steps per block
    for (int s = wid; s < steps; s += 8) {
        int n0 = (ks * steps + s) * 32;
        v16bf a[2], bb[2];
#pragma unroll
        for (int t = 0; t < 2; ++t) {
            a[t]  = av[t] ? load_a_frag(Arow[t] + n0, hi)  : zero_frag();
            bb[t] = bv[t] ? load_b_frag(Brow[t] + n0, k0b) : zero_frag();
        }
#pragma unroll
        for (int m2 = 0; m2 < 2; ++m2)
#pragma unroll
            for (int n2 = 0; n2 < 2; ++n2)
                acc[m2][n2] = __builtin_amdgcn_wmma_f32_16x16x32_bf16(
                    false, a[m2], false, bb[n2], (short)0, acc[m2][n2], false, false);
    }
#pragma unroll
    for (int m2 = 0; m2 < 2; ++m2)
#pragma unroll
        for (int n2 = 0; n2 < 2; ++n2)
#pragma unroll
            for (int i = 0; i < 8; ++i) {
                int r = m2 * 16 + i + hi;
                int c = n2 * 16 + lm;
                atomicAdd(&attn[(size_t)bh * 1024 + r * 32 + c], acc[m2][n2][i]);
            }
}

// softmax over d (24 valid) with temperature; write bf16 32x32 tiles zero-padded
__global__ void softmax_kernel(const float* __restrict__ a, const float* __restrict__ temp,
                               bf16* __restrict__ ab) {
    int rid = blockIdx.x * 256 + threadIdx.x;   // 32 bh * 32 padded rows
    if (rid >= 32 * 32) return;
    int bh = rid >> 5, cp = rid & 31;
    bf16* o = ab + (size_t)rid * 32;
    if (cp >= 24) { for (int d = 0; d < 32; ++d) o[d] = (bf16)0.0f; return; }
    float t = temp[bh & 7];
    const float* row = a + (size_t)rid * 32;
    float vals[24], m = -1e30f;
    for (int d = 0; d < 24; ++d) { vals[d] = row[d] * t; m = fmaxf(m, vals[d]); }
    float s = 0.f;
    for (int d = 0; d < 24; ++d) { vals[d] = __expf(vals[d] - m); s += vals[d]; }
    float inv = 1.0f / s;
    for (int d = 0; d < 24; ++d) o[d] = (bf16)(vals[d] * inv);
    for (int d = 24; d < 32; ++d) o[d] = (bf16)0.0f;
}

// out_t[p,c] = sum_d attn[bh,c%24,d] * v[b, 48+g*24+d, p]   (K=24, bandwidth bound)
__global__ void attn_v_kernel(const bf16* __restrict__ ab, const bf16* __restrict__ kv,
                              bf16* __restrict__ out) {
    long idx = (long)blockIdx.x * 256 + threadIdx.x;  // NB*HW*CDIM
    int  c  = (int)(idx % CDIM);
    long bp = idx / CDIM;
    int  p  = (int)(bp % HW);
    int  b  = (int)(bp / HW);
    int  h  = c / 24, ci = c % 24, g = h >> 2;
    const bf16* arow = ab + ((size_t)(b * 8 + h) * 32 + ci) * 32;
    const bf16* vcol = kv + ((size_t)b * CKV + CK + g * 24) * HW + p;
    float acc = 0.f;
#pragma unroll
    for (int d = 0; d < 24; ++d) acc += (float)arow[d] * (float)vcol[(size_t)d * HW];
    out[idx] = (bf16)acc;
}

// ---------- host ----------
extern "C" void kernel_launch(void* const* d_in, const int* in_sizes, int n_in,
                              void* d_out, int out_size, void* d_ws, size_t ws_size,
                              hipStream_t stream) {
    const float* x      = (const float*)d_in[0];
    const float* ln_w   = (const float*)d_in[1];
    const float* ln_b   = (const float*)d_in[2];
    const float* q_w    = (const float*)d_in[3];
    const float* q_dw   = (const float*)d_in[4];
    const float* kv_w   = (const float*)d_in[5];
    const float* kv_dw  = (const float*)d_in[6];
    const float* proj_w = (const float*)d_in[7];
    const float* temp   = (const float*)d_in[8];
    float* out = (float*)d_out;

    char* ws = (char*)d_ws;
    size_t off = 0;
    auto alloc = [&](size_t bytes) { size_t o = off; off += (bytes + 255) & ~(size_t)255; return o; };
    const size_t BIG  = (size_t)NB * HW * CDIM * 2;  // 100.7 MB
    const size_t MED  = (size_t)NB * HW * CKV  * 2;  // 50.3 MB
    bf16* BUF0  = (bf16*)(ws + alloc(BIG));          // xn_t -> q_dw_t -> out_t
    bf16* BUF1  = (bf16*)(ws + alloc(BIG));          // q_conv_t -> q_cf
    bf16* BUF2  = (bf16*)(ws + alloc(MED));          // kv_conv_t -> kv_cf
    bf16* BUF3  = (bf16*)(ws + alloc(MED));          // kv_dw_t
    bf16* WQ    = (bf16*)(ws + alloc((size_t)CDIM * CDIM * 2));
    bf16* WKV   = (bf16*)(ws + alloc((size_t)CKV  * CDIM * 2));
    bf16* WP    = (bf16*)(ws + alloc((size_t)CDIM * CDIM * 2));
    float* RNQ  = (float*)(ws + alloc((size_t)NB * CDIM * 4));
    float* RNK  = (float*)(ws + alloc((size_t)NB * CK * 4));
    float* ATTN = (float*)(ws + alloc(32 * 1024 * 4));
    bf16*  ATTNB= (bf16*) (ws + alloc(32 * 1024 * 2));

    // 1) weights -> bf16
    cvt_bf_kernel<<<(CDIM * CDIM + 255) / 256, 256, 0, stream>>>(q_w, WQ, CDIM * CDIM);
    cvt_bf_kernel<<<(CKV * CDIM + 255) / 256, 256, 0, stream>>>(kv_w, WKV, CKV * CDIM);
    cvt_bf_kernel<<<(CDIM * CDIM + 255) / 256, 256, 0, stream>>>(proj_w, WP, CDIM * CDIM);
    // 2) LayerNorm -> channel-last bf16
    ln_kernel<<<NB * HW / 256, 256, 0, stream>>>(x, ln_w, ln_b, BUF0);
    // 3) conv1x1 (WMMA, 2x2 blocked): q and kv
    gemm_fwd_kernel<<<(NB * HW / 32) * (CDIM / 32) / 8, 256, 0, stream>>>(BUF0, WQ, BUF1, CDIM, CDIM / 32);
    gemm_fwd_kernel<<<(NB * HW / 32) * (CKV / 32) / 8, 256, 0, stream>>>(BUF0, WKV, BUF2, CKV, CKV / 32);
    // 4) depthwise 3x3 (q output overwrites xn buffer)
    dwconv_kernel<<<(int)((size_t)NB * HW * CDIM / 256), 256, 0, stream>>>(BUF1, q_dw, BUF0, CDIM);
    dwconv_kernel<<<(int)((size_t)NB * HW * CKV / 256), 256, 0, stream>>>(BUF2, kv_dw, BUF3, CKV);
    // 5) transpose to channel-first for the pixel-K attention GEMM
    transpose_kernel<<<NB * (HW / 32) * (CDIM / 32), 256, 0, stream>>>(BUF0, BUF1, CDIM);
    transpose_kernel<<<NB * (HW / 32) * (CKV / 32), 256, 0, stream>>>(BUF3, BUF2, CKV);
    // 6) l2 row norms + scaling (q: all 192 ch; k: first 48 ch of kv)
    rownorm_kernel<<<NB * CDIM, 256, 0, stream>>>(BUF1, RNQ, CDIM, CDIM);
    rownorm_kernel<<<NB * CK, 256, 0, stream>>>(BUF2, RNK, CK, CKV);
    scale_rows_kernel<<<(int)((size_t)NB * CDIM * HW / 256), 256, 0, stream>>>(BUF1, RNQ, CDIM, CDIM);
    scale_rows_kernel<<<(int)((size_t)NB * CK * HW / 256), 256, 0, stream>>>(BUF2, RNK, CK, CKV);
    // 7) attn = q.k^T (WMMA split-K with f32 atomics)
    zerof_kernel<<<(32 * 1024 + 255) / 256, 256, 0, stream>>>(ATTN, 32 * 1024);
    qk_kernel<<<32 * KSPLIT, 256, 0, stream>>>(BUF1, BUF2, ATTN);
    // 8) softmax(+temperature) -> bf16, zero-padded to 32x32
    softmax_kernel<<<4, 256, 0, stream>>>(ATTN, temp, ATTNB);
    // 9) out_t = attn @ v (K=24, channel-last bf16; reuses BUF0)
    attn_v_kernel<<<(int)((size_t)NB * HW * CDIM / 256), 256, 0, stream>>>(ATTNB, BUF2, BUF0);
    // 10) proj (WMMA, 2x2 blocked) -> f32 channel-first d_out
    gemm_proj_kernel<<<(NB * HW / 32) * (CDIM / 32) / 8, 256, 0, stream>>>(WP, BUF0, out);

    (void)in_sizes; (void)n_in; (void)out_size; (void)ws_size;
}
